// EncoderLayer_63883343560772
// MI455X (gfx1250) — matrix operand order repaired and checked by reading, not compiled, not run on hardware
//
#include <hip/hip_runtime.h>
#include <hip/hip_bf16.h>
#include <math.h>
#include <stdint.h>

// ---------------------------------------------------------------------------
// Types for CDNA5 WMMA (gfx1250, wave32)
// ---------------------------------------------------------------------------
typedef __bf16 bf16_t;
typedef __attribute__((ext_vector_type(16))) __bf16 v16bf;
typedef __attribute__((ext_vector_type(8)))  float  v8f;
typedef __attribute__((ext_vector_type(4)))  unsigned int u32x4;
typedef __attribute__((ext_vector_type(8)))  int i32x8;
typedef __attribute__((ext_vector_type(4)))  int i32x4;

// Tensor Data Mover availability (5-arg on ROCm7.2/clang-22, 6-arg on clang-23+)
#if __has_builtin(__builtin_amdgcn_tensor_load_to_lds)
#define GEMM_USE_TDM 1
#else
#define GEMM_USE_TDM 0
#endif

union BF16Frag {
    v16bf  v;
    bf16_t e[16];
    uint4  q[2];
};

__device__ __forceinline__ bf16_t f2bf(float f) {
    union { float f; unsigned u; } v;
    v.f = f;
    unsigned r = (v.u + 0x7fffu + ((v.u >> 16) & 1u)) >> 16;  // RNE
    unsigned short s = (unsigned short)r;
    bf16_t b;
    __builtin_memcpy(&b, &s, sizeof(b));
    return b;
}

// ---------------------------------------------------------------------------
// f32 -> bf16 convert / zero
// ---------------------------------------------------------------------------
__global__ __launch_bounds__(256) void f32_to_bf16_kernel(const float* __restrict__ src,
                                                          bf16_t* __restrict__ dst, int n) {
    int i = blockIdx.x * 256 + threadIdx.x;
    if (i < n) dst[i] = f2bf(src[i]);
}

__global__ __launch_bounds__(256) void zero_f32_kernel(float* __restrict__ p, int n) {
    int i = blockIdx.x * 256 + threadIdx.x;
    if (i < n) p[i] = 0.0f;
}

// ---------------------------------------------------------------------------
// Tiled bf16 WMMA GEMM: C[M,N] = act(A[M,K] @ B[K,N] + bias) (+resid)
// Block tile 128x128, K-step 32, double-buffered LDS.
// 8 waves (2x4), wave tile 64x32 -> 8 v_wmma_f32_16x16x32_bf16 per K-step.
// A tile staged by the Tensor Data Mover (wave 0 issues tensor_load_to_lds,
// waits with s_wait_tensorcnt) while all threads prefetch+transpose B.
// relu applied BEFORE residual add (matches reference FFN: relu(..)+S1).
// ---------------------------------------------------------------------------
template <bool RELU, bool RESID, bool OUTBF>
__global__ __launch_bounds__(256) void gemm_bf16_wmma(
    const bf16_t* __restrict__ A, const bf16_t* __restrict__ Bm,
    const float* __restrict__ bias, const float* __restrict__ resid,
    float* __restrict__ Cf, bf16_t* __restrict__ Cb,
    int N, int Kd)
{
    __shared__ __align__(16) bf16_t As[2][128 * 32];   // [row][k]
    __shared__ __align__(16) bf16_t Bs[2][128 * 32];   // transposed: [n][k]

    const int tid   = threadIdx.x;
    const int lane  = tid & 31;
    const int wave  = tid >> 5;
    const int m0    = blockIdx.y * 128;
    const int n0    = blockIdx.x * 128;
    const int waveM = (wave >> 2) * 64;
    const int waveN = (wave & 3) * 32;

    const int arow  = tid >> 1;        // 0..127   (manual A path)
    const int ahalf = tid & 1;
    const int bk    = tid >> 3;        // 0..31
    const int bn    = (tid & 7) * 16;  // 0..112

    v8f acc[4][2];
    for (int i = 0; i < 4; ++i)
        for (int j = 0; j < 2; ++j)
            for (int r = 0; r < 8; ++r) acc[i][j][r] = 0.0f;

    BF16Frag bstage;
#if !GEMM_USE_TDM
    uint4 astage0, astage1;
#endif

    auto loadB = [&](int t) {
        const uint4* src =
            reinterpret_cast<const uint4*>(Bm + (size_t)(t * 32 + bk) * N + n0 + bn);
        bstage.q[0] = src[0];
        bstage.q[1] = src[1];
    };
    auto commitB = [&](int buf) {
        for (int i = 0; i < 16; ++i) Bs[buf][(bn + i) * 32 + bk] = bstage.e[i];
    };

#if GEMM_USE_TDM
    // TDM: one 2D-tile descriptor per K-step; wave 0 issues, others keep going.
    auto issueA = [&](int t, int buf) {
        if (wave == 0) {
            const bf16_t* gp = A + (size_t)m0 * Kd + t * 32;
            const unsigned long long ga = (unsigned long long)(uintptr_t)gp;
            const unsigned ldsOff = (unsigned)(uintptr_t)&As[buf][0];
            u32x4 g0;
            g0[0] = 1u;                                           // count=1
            g0[1] = ldsOff;                                       // lds_addr
            g0[2] = (unsigned)(ga & 0xffffffffull);               // global_addr lo
            g0[3] = (unsigned)((ga >> 32) & 0x01ffffffull) | (2u << 30);  // hi | type=2
            const unsigned td0 = (unsigned)Kd;                    // tensor_dim0
            const unsigned td1 = 1u << 20;                        // tensor_dim1 (big)
            const unsigned long long st0 = (unsigned long long)(unsigned)Kd;  // row stride
            i32x8 g1;
            g1[0] = (int)0x00010000u;                             // data_size = 2B
            g1[1] = (int)((td0 & 0xffffu) << 16);
            g1[2] = (int)((td0 >> 16) | ((td1 & 0xffffu) << 16));
            g1[3] = (int)((td1 >> 16) | (32u << 16));             // tile_dim0 = 32
            g1[4] = 128;                                          // tile_dim1 = 128
            g1[5] = (int)(st0 & 0xffffffffull);
            g1[6] = (int)((st0 >> 32) & 0xffffull);
            g1[7] = 0;
            i32x4 z4 = {0, 0, 0, 0};
#if defined(__clang_major__) && __clang_major__ >= 23
            i32x8 z8 = {0, 0, 0, 0, 0, 0, 0, 0};
            __builtin_amdgcn_tensor_load_to_lds(g0, g1, z4, z4, z8, 0);
#else
            __builtin_amdgcn_tensor_load_to_lds(g0, g1, z4, z4, 0);
#endif
        }
    };
    auto waitA = [&]() {
        if (wave == 0) __builtin_amdgcn_s_wait_tensorcnt(0);
    };
#else
    auto loadA = [&](int t) {
        const uint4* src =
            reinterpret_cast<const uint4*>(A + (size_t)(m0 + arow) * Kd + t * 32 + ahalf * 16);
        astage0 = src[0];
        astage1 = src[1];
    };
    auto commitA = [&](int buf) {
        uint4* dst = reinterpret_cast<uint4*>(&As[buf][arow * 32 + ahalf * 16]);
        dst[0] = astage0;
        dst[1] = astage1;
    };
#endif

    auto compute = [&](int buf) {
        BF16Frag bfr[2];
        for (int j = 0; j < 2; ++j) {
            const int n = waveN + j * 16 + (lane & 15);
            const uint4* p = reinterpret_cast<const uint4*>(&Bs[buf][n * 32 + (lane >> 4) * 16]);
            bfr[j].q[0] = p[0];
            bfr[j].q[1] = p[1];
        }
        for (int i = 0; i < 4; ++i) {
            BF16Frag afr;
            const int row = waveM + i * 16 + (lane & 15);
            const int kb  = (lane >> 4) * 16;
            const uint4* p0 = reinterpret_cast<const uint4*>(&As[buf][row * 32 + kb]);
            const uint4* p1 = reinterpret_cast<const uint4*>(&As[buf][row * 32 + kb + 8]);
            afr.q[0] = p0[0];
            afr.q[1] = p1[0];
            for (int j = 0; j < 2; ++j) {
                acc[i][j] = __builtin_amdgcn_wmma_f32_16x16x32_bf16(
                    false, afr.v, false, bfr[j].v, (short)0, acc[i][j], false, false);
            }
        }
    };

    const int T = Kd >> 5;

    // prologue: stage tile 0 into buffer 0
    loadB(0);
#if GEMM_USE_TDM
    issueA(0, 0);
    commitB(0);
    waitA();
#else
    loadA(0);
    commitB(0);
    commitA(0);
#endif
    __syncthreads();

    for (int t = 0; t < T; ++t) {
        const int cur = t & 1;
        const int nxt = cur ^ 1;
        if (t + 1 < T) {
            loadB(t + 1);
#if GEMM_USE_TDM
            issueA(t + 1, nxt);
#else
            loadA(t + 1);
#endif
        }
        compute(cur);
        if (t + 1 < T) {
            commitB(nxt);
#if GEMM_USE_TDM
            waitA();
#else
            commitA(nxt);
#endif
        }
        __syncthreads();
    }

    // Epilogue (branch-free: all policy is template-static, bias always present).
    // C/D layout: VGPR r -> M = r + 8*(lane>>4); N = lane&15.
    float bcol[2];
    for (int j = 0; j < 2; ++j) bcol[j] = bias[n0 + waveN + j * 16 + (lane & 15)];
    for (int i = 0; i < 4; ++i) {
        for (int j = 0; j < 2; ++j) {
            const int colb = n0 + waveN + j * 16 + (lane & 15);
            const int rowb = m0 + waveM + i * 16 + ((lane >> 4) ? 8 : 0);
            for (int r = 0; r < 8; ++r) {
                const int row = rowb + r;
                float v = acc[i][j][r] + bcol[j];
                if (RELU) v = fmaxf(v, 0.0f);
                if (RESID) v += resid[(size_t)row * N + colb];
                const size_t o = (size_t)row * N + colb;
                if (OUTBF) Cb[o] = f2bf(v);
                else       Cf[o] = v;
            }
        }
    }
}

// ---------------------------------------------------------------------------
// Stockham radix-2 complex FFT, N=4096, in LDS. 256 threads, 8 bfly/thr/stage.
// Result ends in (reA, imA) after 12 stages. sgn = -1 fwd, +1 inverse (unnorm).
// ---------------------------------------------------------------------------
__device__ __forceinline__ void fft4096_shared(float* reA, float* imA,
                                               float* reB, float* imB,
                                               int tid, float sgn) {
    float* sr = reA; float* si = imA; float* dr = reB; float* di = imB;
    int l = 2048;
    int mshift = 0;
    for (int stage = 0; stage < 12; ++stage) {
        const int m = 1 << mshift;
        for (int t = 0; t < 8; ++t) {
            const int idx = tid + t * 256;          // 0..2047
            const int j   = idx >> mshift;
            const int k   = idx & (m - 1);
            const float ang = sgn * 3.14159265358979323846f * (float)j / (float)l;
            float c, s;
            __sincosf(ang, &s, &c);
            const int i0 = k + j * m;
            const float ar = sr[i0],         ai = si[i0];
            const float br = sr[i0 + l * m], bi = si[i0 + l * m];
            const float er = ar + br, ei = ai + bi;
            const float orr = ar - br, oi = ai - bi;
            const int o0 = k + 2 * j * m;
            dr[o0]     = er;
            di[o0]     = ei;
            dr[o0 + m] = c * orr - s * oi;
            di[o0 + m] = s * orr + c * oi;
        }
        __syncthreads();
        float* tp;
        tp = sr; sr = dr; dr = tp;
        tp = si; si = di; di = tp;
        l >>= 1;
        ++mshift;
    }
}

// One block per (batch, channel): pack z = q + i*k, FFT, unpack Hermitian,
// cross spectrum Q*conj(K), accumulate into per-batch spectrum (linearity of
// irfft lets us average spectra instead of per-channel correlations).
__global__ __launch_bounds__(256) void fft_cross_accum(
    const float* __restrict__ Q, const float* __restrict__ Kq,
    float* __restrict__ Sre, float* __restrict__ Sim, int L, int D)
{
    __shared__ float reA[4096], imA[4096], reB[4096], imB[4096];
    const int tid = threadIdx.x;
    const int b = blockIdx.x / D;
    const int c = blockIdx.x % D;
    const size_t base = (size_t)b * L * D + c;
    for (int t = 0; t < 16; ++t) {
        const int l = tid + t * 256;
        reA[l] = Q[base + (size_t)l * D];
        imA[l] = Kq[base + (size_t)l * D];
    }
    __syncthreads();
    fft4096_shared(reA, imA, reB, imB, tid, -1.0f);
    for (int t = 0; t < 16; ++t) {
        const int f  = tid + t * 256;
        const int fr = (4096 - f) & 4095;
        const float a = reA[f],  bb = imA[f];
        const float cc = reA[fr], dd = imA[fr];
        // Z = Qf + i*Kf  =>  Qf = (Z(f)+conj(Z(-f)))/2 ; Kf = (Z(f)-conj(Z(-f)))/(2i)
        const float qr = 0.5f * (a + cc),  qi = 0.5f * (bb - dd);
        const float kr = 0.5f * (bb + dd), ki = 0.5f * (cc - a);
        const float xr = qr * kr + qi * ki;   // Q * conj(K)
        const float xi = qi * kr - qr * ki;
        atomicAdd(&Sre[b * 4096 + f], xr);
        atomicAdd(&Sim[b * 4096 + f], xi);
    }
}

// One block per batch: inverse FFT of averaged cross spectrum -> mean_corr.
__global__ __launch_bounds__(256) void ifft_mean(
    const float* __restrict__ Sre, const float* __restrict__ Sim,
    float* __restrict__ mc, float scale)
{
    __shared__ float reA[4096], imA[4096], reB[4096], imB[4096];
    const int tid = threadIdx.x;
    const int b = blockIdx.x;
    for (int t = 0; t < 16; ++t) {
        const int f = tid + t * 256;
        reA[f] = Sre[b * 4096 + f];
        imA[f] = Sim[b * 4096 + f];
    }
    __syncthreads();
    fft4096_shared(reA, imA, reB, imB, tid, +1.0f);
    for (int t = 0; t < 16; ++t) {
        const int l = tid + t * 256;
        mc[b * 4096 + l] = reA[l] * scale;
    }
}

// ---------------------------------------------------------------------------
// top-k (k=8) + softmax over a 4096 row; destructive on mc (scratch).
// ---------------------------------------------------------------------------
__global__ __launch_bounds__(256) void topk_softmax(
    float* __restrict__ mc, float* __restrict__ p, int* __restrict__ delay,
    int L, int k)
{
    __shared__ float sval[256];
    __shared__ int   sidx[256];
    __shared__ float wts[8];
    __shared__ int   dls[8];
    const int b = blockIdx.x;
    float* row = mc + (size_t)b * L;
    for (int j = 0; j < k; ++j) {
        float best = -3.402823466e38f;
        int bi = 0;
        for (int i = threadIdx.x; i < L; i += 256) {
            float v = row[i];
            if (v > best) { best = v; bi = i; }
        }
        sval[threadIdx.x] = best;
        sidx[threadIdx.x] = bi;
        __syncthreads();
        for (int off = 128; off > 0; off >>= 1) {
            if ((int)threadIdx.x < off) {
                if (sval[threadIdx.x + off] > sval[threadIdx.x]) {
                    sval[threadIdx.x] = sval[threadIdx.x + off];
                    sidx[threadIdx.x] = sidx[threadIdx.x + off];
                }
            }
            __syncthreads();
        }
        if (threadIdx.x == 0) {
            wts[j] = sval[0];
            dls[j] = sidx[0];
            row[sidx[0]] = -3.402823466e38f;
        }
        __syncthreads();
    }
    if (threadIdx.x == 0) {
        float mx = wts[0];
        for (int j = 1; j < k; ++j) mx = fmaxf(mx, wts[j]);
        float e[8], ssum = 0.0f;
        for (int j = 0; j < k; ++j) { e[j] = __expf(wts[j] - mx); ssum += e[j]; }
        for (int j = 0; j < k; ++j) {
            p[b * k + j]     = e[j] / ssum;
            delay[b * k + j] = dls[j];
        }
    }
}

// ---------------------------------------------------------------------------
// out[b,l,:] = sum_j p[j] * V[b,(l+delay_j) mod L,:]   (emit bf16 for Wo GEMM)
// ---------------------------------------------------------------------------
__global__ __launch_bounds__(256) void gather_weighted(
    const float* __restrict__ V, const float* __restrict__ p,
    const int* __restrict__ delay, bf16_t* __restrict__ out,
    int L, int D, int k)
{
    __shared__ float sp[8];
    __shared__ int   sd[8];
    const size_t idx = (size_t)blockIdx.x * 256 + threadIdx.x;
    const int d = (int)(idx % D);
    const size_t bl = idx / D;
    const int l = (int)(bl % L);
    const int b = (int)(bl / L);
    if ((int)threadIdx.x < k) {
        sp[threadIdx.x] = p[b * k + threadIdx.x];
        sd[threadIdx.x] = delay[b * k + threadIdx.x];
    }
    __syncthreads();
    float s = 0.0f;
    for (int j = 0; j < k; ++j) {
        const int ls = (l + sd[j]) & (L - 1);
        s += sp[j] * V[((size_t)b * L + ls) * D + d];
    }
    out[idx] = f2bf(s);
}

// ---------------------------------------------------------------------------
// Series decomposition: out = h - avgpool_same(h, 25) along L.
// Optionally emits a bf16 copy (FFN GEMM input).
// ---------------------------------------------------------------------------
__global__ __launch_bounds__(256) void decomp_pool25(
    const float* __restrict__ h, float* __restrict__ outF,
    bf16_t* __restrict__ outB, int L, int D)
{
    const size_t idx = (size_t)blockIdx.x * 256 + threadIdx.x;
    const int d = (int)(idx % D);
    const size_t bl = idx / D;
    const int l = (int)(bl % L);
    const int b = (int)(bl / L);
    int lo = l - 12; if (lo < 0) lo = 0;
    int hi = l + 12; if (hi > L - 1) hi = L - 1;
    const float* col = h + (size_t)b * L * D + d;
    float s = 0.0f;
    for (int i = lo; i <= hi; ++i) s += col[(size_t)i * D];
    const float v = col[(size_t)l * D] - s / (float)(hi - lo + 1);
    outF[idx] = v;
    if (outB) outB[idx] = f2bf(v);
}

// ---------------------------------------------------------------------------
// Orchestration
// ---------------------------------------------------------------------------
extern "C" void kernel_launch(void* const* d_in, const int* in_sizes, int n_in,
                              void* d_out, int out_size, void* d_ws, size_t ws_size,
                              hipStream_t stream) {
    (void)in_sizes; (void)n_in; (void)out_size; (void)ws_size;

    const float* x  = (const float*)d_in[0];
    const float* Wq = (const float*)d_in[1];
    const float* bq = (const float*)d_in[2];
    const float* Wk = (const float*)d_in[3];
    const float* bk = (const float*)d_in[4];
    const float* Wv = (const float*)d_in[5];
    const float* bv = (const float*)d_in[6];
    const float* Wo = (const float*)d_in[7];
    const float* bo = (const float*)d_in[8];
    const float* W1 = (const float*)d_in[9];
    const float* b1 = (const float*)d_in[10];
    const float* W2 = (const float*)d_in[11];
    const float* b2 = (const float*)d_in[12];
    float* out = (float*)d_out;

    const int Bc = 4, L = 4096, D = 512, DFF = 2048, TK = 8;
    const int Mrows = Bc * L;  // 16384

    // --- workspace carve (with reuse) ---
    char* base = (char*)d_ws;
    size_t off = 0;
    auto alloc = [&](size_t bytes) -> void* {
        off = (off + 255) & ~(size_t)255;
        void* pp = base + off;
        off += bytes;
        return pp;
    };
    bf16_t* xb   = (bf16_t*)alloc((size_t)Mrows * D * sizeof(bf16_t));
    bf16_t* Wqb  = (bf16_t*)alloc((size_t)D * D * sizeof(bf16_t));
    bf16_t* Wkb  = (bf16_t*)alloc((size_t)D * D * sizeof(bf16_t));
    bf16_t* Wvb  = (bf16_t*)alloc((size_t)D * D * sizeof(bf16_t));
    bf16_t* Wob  = (bf16_t*)alloc((size_t)D * D * sizeof(bf16_t));
    bf16_t* W1b  = (bf16_t*)alloc((size_t)D * DFF * sizeof(bf16_t));
    bf16_t* W2b  = (bf16_t*)alloc((size_t)DFF * D * sizeof(bf16_t));
    float*  Qf   = (float*)alloc((size_t)Mrows * D * sizeof(float));
    float*  Kf   = (float*)alloc((size_t)Mrows * D * sizeof(float));
    float*  Vf   = (float*)alloc((size_t)Mrows * D * sizeof(float));
    float*  Sre  = (float*)alloc((size_t)2 * Bc * L * sizeof(float));  // Sre|Sim
    float*  Sim  = Sre + (size_t)Bc * L;
    float*  mc   = (float*)alloc((size_t)Bc * L * sizeof(float));
    float*  pW   = (float*)alloc((size_t)Bc * TK * sizeof(float));
    int*    dly  = (int*)alloc((size_t)Bc * TK * sizeof(int));
    bf16_t* gatb = (bf16_t*)alloc((size_t)Mrows * D * sizeof(bf16_t));
    bf16_t* Fb   = (bf16_t*)alloc((size_t)Mrows * DFF * sizeof(bf16_t));
    // reuse: h <- Qf, S1 <- Kf, S1(bf16) <- gatb, h2 <- Vf
    float*  Hf  = Qf;
    float*  S1f = Kf;
    bf16_t* S1b = gatb;
    float*  H2f = Vf;

    // --- bf16 conversions ---
    int n;
    n = Mrows * D;  f32_to_bf16_kernel<<<(n + 255) / 256, 256, 0, stream>>>(x,  xb,  n);
    n = D * D;      f32_to_bf16_kernel<<<(n + 255) / 256, 256, 0, stream>>>(Wq, Wqb, n);
    n = D * D;      f32_to_bf16_kernel<<<(n + 255) / 256, 256, 0, stream>>>(Wk, Wkb, n);
    n = D * D;      f32_to_bf16_kernel<<<(n + 255) / 256, 256, 0, stream>>>(Wv, Wvb, n);
    n = D * D;      f32_to_bf16_kernel<<<(n + 255) / 256, 256, 0, stream>>>(Wo, Wob, n);
    n = D * DFF;    f32_to_bf16_kernel<<<(n + 255) / 256, 256, 0, stream>>>(W1, W1b, n);
    n = DFF * D;    f32_to_bf16_kernel<<<(n + 255) / 256, 256, 0, stream>>>(W2, W2b, n);

    // --- Q/K/V projections (WMMA) ---
    dim3 gD(D / 128, Mrows / 128);
    gemm_bf16_wmma<false, false, false><<<gD, 256, 0, stream>>>(xb, Wqb, bq, nullptr, Qf, nullptr, D, D);
    gemm_bf16_wmma<false, false, false><<<gD, 256, 0, stream>>>(xb, Wkb, bk, nullptr, Kf, nullptr, D, D);
    gemm_bf16_wmma<false, false, false><<<gD, 256, 0, stream>>>(xb, Wvb, bv, nullptr, Vf, nullptr, D, D);

    // --- autocorrelation: spectra, average, single irfft per batch ---
    n = 2 * Bc * L;
    zero_f32_kernel<<<(n + 255) / 256, 256, 0, stream>>>(Sre, n);
    fft_cross_accum<<<Bc * D, 256, 0, stream>>>(Qf, Kf, Sre, Sim, L, D);
    ifft_mean<<<Bc, 256, 0, stream>>>(Sre, Sim, mc, 1.0f / (4096.0f * 512.0f));

    // --- top-8 delays + softmax, weighted circular gather of V ---
    topk_softmax<<<Bc, 256, 0, stream>>>(mc, pW, dly, L, TK);
    gather_weighted<<<(Mrows * D) / 256, 256, 0, stream>>>(Vf, pW, dly, gatb, L, D, TK);

    // --- output projection + residual; decomposition 1 ---
    gemm_bf16_wmma<false, true, false><<<gD, 256, 0, stream>>>(gatb, Wob, bo, x, Hf, nullptr, D, D);
    decomp_pool25<<<(Mrows * D) / 256, 256, 0, stream>>>(Hf, S1f, S1b, L, D);

    // --- FFN (WMMA) + residual; decomposition 2 -> output ---
    dim3 gF(DFF / 128, Mrows / 128);
    gemm_bf16_wmma<true, false, true><<<gF, 256, 0, stream>>>(S1b, W1b, b1, nullptr, nullptr, Fb, DFF, D);
    gemm_bf16_wmma<true, true, false><<<gD, 256, 0, stream>>>(Fb, W2b, b2, S1f, H2f, nullptr, D, DFF);
    decomp_pool25<<<(Mrows * D) / 256, 256, 0, stream>>>(H2f, out, nullptr, L, D);
}
